// TreeNetCell_88210038325568
// MI455X (gfx1250) — matrix operand order repaired and checked by reading, not compiled, not run on hardware
//
#include <hip/hip_runtime.h>
#include <hip/hip_bf16.h>

typedef __attribute__((ext_vector_type(16))) __bf16 v16bf;
typedef __attribute__((ext_vector_type(8)))  float  v8f;

#define B_ROWS 32768
#define CCH    8
#define HH     128
#define KK     1024      // C*H
#define NTOT   1152      // 1024 gate cols + 128 iou cols
#define MT     64        // batch rows per workgroup (main kernel)
#define APAD   8
#define AST    (KK + APAD)   // padded LDS row stride (elements)

// ---------------- workspace layout ----------------
// [0)            Wt      : bf16 [1152][1024]  (N-major W_f | W_aggr)   2,359,296 B
// [2359296)      bias    : f32  [1152]        (b_f | b_aggr)               4,608 B
// [2363904)      WtF     : bf16 [128][128]    (N-major W_fin)             32,768 B
// [2396672)      f_input : f32  [32768][128]                          16,777,216 B
#define WS_BIAS 2359296
#define WS_WTF  2363904
#define WS_FIN  2396672

// Pack W_f (K x 1024) and W_aggr (K x 128) into N-major bf16 Wt[n][k].
__global__ void pack_w_kernel(const float* __restrict__ W_f,
                              const float* __restrict__ W_aggr,
                              __bf16* __restrict__ Wt) {
    int idx = blockIdx.x * 256 + threadIdx.x;      // 1152*1024 threads
    int n = idx >> 10;
    int k = idx & 1023;
    float v = (n < 1024) ? W_f[(size_t)k * 1024 + n]
                         : W_aggr[(size_t)k * 128 + (n - 1024)];
    Wt[idx] = (__bf16)v;
}

// Pack W_fin (128x128) N-major bf16 + combined bias vector.
__global__ void pack_misc_kernel(const float* __restrict__ W_fin,
                                 const float* __restrict__ b_f,
                                 const float* __restrict__ b_aggr,
                                 __bf16* __restrict__ WtF,
                                 float* __restrict__ bias) {
    int idx = blockIdx.x * 256 + threadIdx.x;
    if (idx < 128 * 128) {
        int n = idx >> 7, k = idx & 127;
        WtF[idx] = (__bf16)W_fin[k * 128 + n];
    } else if (idx < 128 * 128 + NTOT) {
        int n = idx - 128 * 128;
        bias[n] = (n < 1024) ? b_f[n] : b_aggr[n - 1024];
    }
}

// f_input = (x_embs @ W_fin + b_fin) * x_mask   via bf16 WMMA, f32 accum.
__global__ void finput_kernel(const float* __restrict__ x,
                              const float* __restrict__ mask,
                              const __bf16* __restrict__ WtF,
                              const float* __restrict__ b_fin,
                              float* __restrict__ f_out) {
    const int lane = threadIdx.x & 31;
    const int w    = threadIdx.x >> 5;
    const int lm   = lane & 15;
    const int lk   = lane >> 4;                  // 0/1
    const size_t m_base = (size_t)blockIdx.x * 128 + w * 16;

    v8f acc[8];
    const v8f z = {0.f,0.f,0.f,0.f,0.f,0.f,0.f,0.f};
#pragma unroll
    for (int nt = 0; nt < 8; ++nt) acc[nt] = z;

    const float* Ar = x + (m_base + lm) * 128 + lk * 8;
#pragma unroll
    for (int ks = 0; ks < 128; ks += 32) {
        float4 q0 = *(const float4*)(Ar + ks);
        float4 q1 = *(const float4*)(Ar + ks + 4);
        float4 q2 = *(const float4*)(Ar + ks + 16);
        float4 q3 = *(const float4*)(Ar + ks + 20);
        v16bf a;
        __bf16* ap = (__bf16*)&a;
        ap[0]=(__bf16)q0.x; ap[1]=(__bf16)q0.y; ap[2]=(__bf16)q0.z; ap[3]=(__bf16)q0.w;
        ap[4]=(__bf16)q1.x; ap[5]=(__bf16)q1.y; ap[6]=(__bf16)q1.z; ap[7]=(__bf16)q1.w;
        ap[8]=(__bf16)q2.x; ap[9]=(__bf16)q2.y; ap[10]=(__bf16)q2.z; ap[11]=(__bf16)q2.w;
        ap[12]=(__bf16)q3.x; ap[13]=(__bf16)q3.y; ap[14]=(__bf16)q3.z; ap[15]=(__bf16)q3.w;
#pragma unroll
        for (int nt = 0; nt < 8; ++nt) {
            const __bf16* Br = WtF + (size_t)(nt * 16 + lm) * 128 + ks + lk * 8;
            v16bf b;
            ((uint4*)&b)[0] = *(const uint4*)(Br);
            ((uint4*)&b)[1] = *(const uint4*)(Br + 16);
            acc[nt] = __builtin_amdgcn_wmma_f32_16x16x32_bf16(
                false, a, false, b, (short)0, acc[nt], false, false);
        }
    }
#pragma unroll
    for (int nt = 0; nt < 8; ++nt) {
        int h = nt * 16 + lm;
        float bb = b_fin[h];
#pragma unroll
        for (int j = 0; j < 8; ++j) {
            size_t m = m_base + j + lk * 8;
            f_out[m * 128 + h] = (acc[nt][j] + bb) * mask[m];
        }
    }
}

#define LOAD_AB(kk, A0v, A1v, B0v, B1v)                                  \
    do {                                                                 \
        ((uint4*)&(A0v))[0] = *(const uint4*)(Arow0 + (kk));             \
        ((uint4*)&(A0v))[1] = *(const uint4*)(Arow0 + (kk) + 16);        \
        ((uint4*)&(A1v))[0] = *(const uint4*)(Arow1 + (kk));             \
        ((uint4*)&(A1v))[1] = *(const uint4*)(Arow1 + (kk) + 16);        \
        ((uint4*)&(B0v))[0] = *(const uint4*)(Bp + (kk));                \
        ((uint4*)&(B0v))[1] = *(const uint4*)(Bp + (kk) + 16);           \
        ((uint4*)&(B1v))[0] = *(const uint4*)(Bp + 16 * KK + (kk));      \
        ((uint4*)&(B1v))[1] = *(const uint4*)(Bp + 16 * KK + (kk) + 16); \
    } while (0)

#define DO_WMMA(A0v, A1v, B0v, B1v)                                         \
    do {                                                                    \
        acc[0][0] = __builtin_amdgcn_wmma_f32_16x16x32_bf16(                \
            false, (A0v), false, (B0v), (short)0, acc[0][0], false, false); \
        acc[0][1] = __builtin_amdgcn_wmma_f32_16x16x32_bf16(                \
            false, (A0v), false, (B1v), (short)0, acc[0][1], false, false); \
        acc[1][0] = __builtin_amdgcn_wmma_f32_16x16x32_bf16(                \
            false, (A1v), false, (B0v), (short)0, acc[1][0], false, false); \
        acc[1][1] = __builtin_amdgcn_wmma_f32_16x16x32_bf16(                \
            false, (A1v), false, (B1v), (short)0, acc[1][1], false, false); \
    } while (0)

// Main fused kernel: gather nh -> LDS bf16 A, GEMM vs Wt (1152 cols) in 9
// chunks of 128 with full-K accumulation, fused sigmoid/nc/tanh epilogue.
// 8 waves: 2 M-groups x 4 N-groups; each wave computes a 32x32 tile per chunk
// (2x2 wmma tiles) with register double-buffered A/B fragments.
__global__ void treecell_main_kernel(const float* __restrict__ neighbour_h,
                                     const float* __restrict__ neighbour_c,
                                     const int*   __restrict__ pos,
                                     const __bf16* __restrict__ Wt,
                                     const float* __restrict__ bias,
                                     const float* __restrict__ f_input,
                                     float* __restrict__ out_h,
                                     float* __restrict__ out_c) {
    extern __shared__ char smem[];
    __bf16* A_s   = (__bf16*)smem;                        // [MT][AST]
    int*    pos_s = (int*)(smem + (size_t)MT * AST * 2);  // [MT][8]

    const int tid = threadIdx.x;
    const size_t rowb0 = (size_t)blockIdx.x * MT;

    // stage pos
    for (int i = tid; i < MT * CCH; i += 256)
        pos_s[i] = pos[rowb0 * CCH + i];
    __syncthreads();

    // gather neighbour_h -> bf16 LDS tile (float4 granularity)
    const float4* nh4 = (const float4*)neighbour_h;
    for (int i = tid; i < MT * 256; i += 256) {   // MT*1024/4 float4 chunks
        int m = i >> 8;
        int r = i & 255;
        int c = r >> 5;
        int j = r & 31;
        float4 v = nh4[(rowb0 + m) * 256 + (size_t)pos_s[m * 8 + c] * 32 + j];
        union { __bf16 b[4]; uint2 u; } p;
        p.b[0]=(__bf16)v.x; p.b[1]=(__bf16)v.y; p.b[2]=(__bf16)v.z; p.b[3]=(__bf16)v.w;
        *(uint2*)&A_s[(size_t)m * AST + c * 128 + j * 4] = p.u;
    }
    __syncthreads();

    const int lane = tid & 31;
    const int w    = tid >> 5;
    const int mg   = (w & 1) * 32;    // wave's M group (32 rows)
    const int ng   = (w >> 1) * 32;   // wave's N group (32 cols of 128 chunk)
    const int lm   = lane & 15;
    const int lk   = lane >> 4;

    const __bf16* Arow0 = &A_s[(size_t)(mg + lm) * AST + lk * 8];
    const __bf16* Arow1 = Arow0 + (size_t)16 * AST;

    const v8f z = {0.f,0.f,0.f,0.f,0.f,0.f,0.f,0.f};
    float c_acc[2][2][8];
#pragma unroll
    for (int ma = 0; ma < 2; ++ma)
#pragma unroll
        for (int nb = 0; nb < 2; ++nb)
#pragma unroll
            for (int j = 0; j < 8; ++j) c_acc[ma][nb][j] = 0.f;

    v8f acc[2][2];
    v16bf A0e, A1e, B0e, B1e, A0o, A1o, B0o, B1o;

    // ---- 8 gate chunks ----
    for (int t = 0; t < 8; ++t) {
        const __bf16* Bp = Wt + (size_t)(t * 128 + ng + lm) * KK + lk * 8;
#pragma unroll
        for (int ma = 0; ma < 2; ++ma)
#pragma unroll
            for (int nb = 0; nb < 2; ++nb) acc[ma][nb] = z;

        LOAD_AB(0, A0e, A1e, B0e, B1e);
        for (int k = 0; k < KK; k += 64) {
            LOAD_AB(k + 32, A0o, A1o, B0o, B1o);
            DO_WMMA(A0e, A1e, B0e, B1e);
            int k2 = (k + 64) & (KK - 1);
            LOAD_AB(k2, A0e, A1e, B0e, B1e);
            DO_WMMA(A0o, A1o, B0o, B1o);
        }

        // fused sigmoid * nc accumulation for this 128-column gate chunk
#pragma unroll
        for (int ma = 0; ma < 2; ++ma)
#pragma unroll
            for (int nb = 0; nb < 2; ++nb) {
                int h = ng + nb * 16 + lm;
                float bb = bias[t * 128 + h];
#pragma unroll
                for (int j = 0; j < 8; ++j) {
                    int m = mg + ma * 16 + lk * 8 + j;
                    size_t row = rowb0 + m;
                    float fi  = f_input[row * HH + h];
                    float val = acc[ma][nb][j] + bb + 2.f * fi;
                    float f   = 1.f / (1.f + __expf(-val));
                    float nc  = neighbour_c[(row * CCH + pos_s[m * 8 + t]) * HH + h];
                    c_acc[ma][nb][j] += f * nc;
                }
            }
    }

    // ---- iou chunk (cols 1024..1151) + final outputs ----
    {
        const __bf16* Bp = Wt + (size_t)(1024 + ng + lm) * KK + lk * 8;
#pragma unroll
        for (int ma = 0; ma < 2; ++ma)
#pragma unroll
            for (int nb = 0; nb < 2; ++nb) acc[ma][nb] = z;

        LOAD_AB(0, A0e, A1e, B0e, B1e);
        for (int k = 0; k < KK; k += 64) {
            LOAD_AB(k + 32, A0o, A1o, B0o, B1o);
            DO_WMMA(A0e, A1e, B0e, B1e);
            int k2 = (k + 64) & (KK - 1);
            LOAD_AB(k2, A0e, A1e, B0e, B1e);
            DO_WMMA(A0o, A1o, B0o, B1o);
        }

#pragma unroll
        for (int ma = 0; ma < 2; ++ma)
#pragma unroll
            for (int nb = 0; nb < 2; ++nb) {
                int h = ng + nb * 16 + lm;
                float bb = bias[1024 + h];
#pragma unroll
                for (int j = 0; j < 8; ++j) {
                    int m = mg + ma * 16 + lk * 8 + j;
                    size_t row = rowb0 + m;
                    float iou = acc[ma][nb][j] + bb;
                    float cc  = c_acc[ma][nb][j];
                    out_h[row * HH + h] = iou * tanhf(cc);
                    out_c[row * HH + h] = cc;
                }
            }
    }
}

extern "C" void kernel_launch(void* const* d_in, const int* in_sizes, int n_in,
                              void* d_out, int out_size, void* d_ws, size_t ws_size,
                              hipStream_t stream) {
    const float* x_embs      = (const float*)d_in[0];
    const float* x_mask      = (const float*)d_in[1];
    const float* neighbour_h = (const float*)d_in[2];
    const float* neighbour_c = (const float*)d_in[3];
    const int*   pos         = (const int*)  d_in[4];
    const float* W_fin       = (const float*)d_in[5];
    const float* b_fin       = (const float*)d_in[6];
    const float* W_aggr      = (const float*)d_in[7];
    const float* b_aggr      = (const float*)d_in[8];
    const float* W_f         = (const float*)d_in[9];
    const float* b_f         = (const float*)d_in[10];

    char* ws = (char*)d_ws;
    __bf16* Wt      = (__bf16*)ws;
    float*  bias    = (float*) (ws + WS_BIAS);
    __bf16* WtF     = (__bf16*)(ws + WS_WTF);
    float*  f_inbuf = (float*) (ws + WS_FIN);

    float* out_h = (float*)d_out;
    float* out_c = out_h + (size_t)B_ROWS * HH;

    pack_w_kernel<<<(NTOT * KK) / 256, 256, 0, stream>>>(W_f, W_aggr, Wt);
    pack_misc_kernel<<<(128 * 128 + NTOT + 255) / 256, 256, 0, stream>>>(
        W_fin, b_f, b_aggr, WtF, bias);
    finput_kernel<<<B_ROWS / 128, 256, 0, stream>>>(
        x_embs, x_mask, WtF, b_fin, f_inbuf);

    size_t shmem = (size_t)MT * AST * 2 + (size_t)MT * CCH * 4;  // 134,144 B
    treecell_main_kernel<<<B_ROWS / MT, 256, shmem, stream>>>(
        neighbour_h, neighbour_c, pos, Wt, bias, f_inbuf, out_h, out_c);
}
